// PointNetMSG_65781719105613
// MI455X (gfx1250) — matrix operand order repaired and checked by reading, not compiled, not run on hardware
//
#include <hip/hip_runtime.h>

// ---------------------------------------------------------------------------
// Types for CDNA5 WMMA (gfx1250, wave32)
// ---------------------------------------------------------------------------
typedef __bf16 bf16_t;
typedef __attribute__((ext_vector_type(16))) __bf16 v16bf;
typedef __attribute__((ext_vector_type(8)))  __bf16 v8bf;
typedef __attribute__((ext_vector_type(8)))  float  v8f;

#define NEG_SLOPE 0.2f
#define BN_EPS 1e-05f

__device__ __forceinline__ bf16_t f2bf(float f) {
    union { float f; unsigned u; } c; c.f = f;
    unsigned u = c.u;
    u += 0x7fffu + ((u >> 16) & 1u);          // round-to-nearest-even
    unsigned short h = (unsigned short)(u >> 16);
    bf16_t r; __builtin_memcpy(&r, &h, 2); return r;
}
__device__ __forceinline__ float bf2f(bf16_t x) {
    unsigned short h; __builtin_memcpy(&h, &x, 2);
    union { unsigned u; float f; } c; c.u = ((unsigned)h) << 16;
    return c.f;
}
__device__ __forceinline__ float lrelu(float x) { return x > 0.f ? x : NEG_SLOPE * x; }

// ---------------------------------------------------------------------------
// Farthest point sampling: one block per batch. dist[] lives in dynamic LDS.
// Matches reference: sel[0]=0; sel[i]=argmax of running min-distance.
// ---------------------------------------------------------------------------
__global__ void fps_kernel(const float* __restrict__ xyz, int N, int S,
                           int* __restrict__ fidx, float* __restrict__ nxyz) {
    extern __shared__ float dist[];
    __shared__ float s_v[256];
    __shared__ int   s_i[256];
    __shared__ int   s_last;
    const int b = blockIdx.x, tid = threadIdx.x;
    const float* base = xyz + (size_t)b * N * 3;
    for (int i = tid; i < N; i += 256) dist[i] = 1e10f;
    if (tid == 0) s_last = 0;
    __syncthreads();
    for (int s = 0; s < S; ++s) {
        int last = s_last;
        float lx = base[last * 3 + 0], ly = base[last * 3 + 1], lz = base[last * 3 + 2];
        if (tid == 0) {
            fidx[b * S + s] = last;
            nxyz[((size_t)b * S + s) * 3 + 0] = lx;
            nxyz[((size_t)b * S + s) * 3 + 1] = ly;
            nxyz[((size_t)b * S + s) * 3 + 2] = lz;
        }
        float bv = -1.f; int bi = 0;
        for (int i = tid; i < N; i += 256) {
            float dx = base[i * 3 + 0] - lx;
            float dy = base[i * 3 + 1] - ly;
            float dz = base[i * 3 + 2] - lz;
            float d  = dx * dx + dy * dy + dz * dz;
            float nd = dist[i]; nd = d < nd ? d : nd; dist[i] = nd;
            if (nd > bv) { bv = nd; bi = i; }
        }
        s_v[tid] = bv; s_i[tid] = bi;
        __syncthreads();
        for (int o = 128; o > 0; o >>= 1) {
            if (tid < o && s_v[tid + o] > s_v[tid]) { s_v[tid] = s_v[tid + o]; s_i[tid] = s_i[tid + o]; }
            __syncthreads();
        }
        if (tid == 0) s_last = s_i[0];
        __syncthreads();
    }
}

// ---------------------------------------------------------------------------
// Ball query + group + concat(gxyz, feats) -> padded bf16 row matrix.
// One wave32 per query point; ballot + prefix popcount collects the first
// `ns` indices within radius in ascending order (== reference sort+truncate).
// ---------------------------------------------------------------------------
__global__ void group_kernel(const float* __restrict__ xyz,   // [B,N,3]
                             const float* __restrict__ feats, // [B,N,C]
                             int C,
                             const float* __restrict__ nxyz,  // [B,S,3]
                             int N, int S, int BS, int ns, float r2, int Kp,
                             bf16_t* __restrict__ out) {      // [BS*ns, Kp]
    __shared__ int sidx[8][128];
    const int w = threadIdx.x >> 5, lane = threadIdx.x & 31;
    const int gs = blockIdx.x * 8 + w;
    if (gs >= BS) return;
    const int b = gs / S;
    const float cx = nxyz[(size_t)gs * 3 + 0];
    const float cy = nxyz[(size_t)gs * 3 + 1];
    const float cz = nxyz[(size_t)gs * 3 + 2];
    const float* xb = xyz + (size_t)b * N * 3;
    int count = 0;
    for (int base = 0; base < N && count < ns; base += 32) {
        int i = base + lane;
        bool in = false;
        if (i < N) {
            float dx = xb[i * 3 + 0] - cx;
            float dy = xb[i * 3 + 1] - cy;
            float dz = xb[i * 3 + 2] - cz;
            in = (dx * dx + dy * dy + dz * dz) < r2;
        }
        unsigned mask = (unsigned)__ballot(in);
        if (in) {
            int pos = count + __popc(mask & ((1u << lane) - 1u));
            if (pos < ns) sidx[w][pos] = i;
        }
        count += __popc(mask);
    }
    if (count > ns) count = ns;
    for (int j = count + lane; j < ns; j += 32) sidx[w][j] = sidx[w][0];
    for (int j = 0; j < ns; ++j) {
        const int id = sidx[w][j];
        const float* fp = feats + ((size_t)b * N + id) * C;
        const float* xp = xb + (size_t)id * 3;
        const size_t rowOff = ((size_t)gs * ns + j) * Kp;
        for (int c = lane; c < Kp; c += 32) {
            float v;
            if      (c == 0) v = xp[0] - cx;
            else if (c == 1) v = xp[1] - cy;
            else if (c == 2) v = xp[2] - cz;
            else if (c < 3 + C) v = fp[c - 3];
            else v = 0.f;
            out[rowOff + c] = f2bf(v);
        }
    }
}

// ---------------------------------------------------------------------------
// Weight fp32 [Cout,Cin] -> bf16 [Cout,Kp] (K zero-padded to mult of 32).
// ---------------------------------------------------------------------------
__global__ void wconv_kernel(const float* __restrict__ Wf, int Cout, int Cin, int Kp,
                             bf16_t* __restrict__ WB) {
    size_t n = (size_t)Cout * Kp;
    for (size_t idx = blockIdx.x * (size_t)blockDim.x + threadIdx.x; idx < n;
         idx += (size_t)gridDim.x * blockDim.x) {
        int k = (int)(idx % Kp);
        size_t o = idx / Kp;
        float v = (k < Cin) ? Wf[o * Cin + k] : 0.f;
        WB[idx] = f2bf(v);
    }
}

// ---------------------------------------------------------------------------
// GEMM:  Y[rows,Np] = X[rows,Kp](bf16) * W[Np,Kp](bf16)^T, fp32 accumulate.
// One wave = one 16x(NT*16) tile with NT compile-time accumulators: the A
// fragment is loaded once per K-step and reused by NT unconditional
// v_wmma_f32_16x16x32_bf16 ops (no divergent guards -> no saveexec/spills).
// Host launches <4> for 64-wide chunks and <2> for the 32-col remainder,
// so the per-layer column count (multiple of 32) is covered exactly.
// A fragment : lane(l16,half) holds X[row0+l16][kb+half*8 .. +7] and [+16..].
// B fragment : lane(l16,half) holds W[col+l16][kb+half*8 .. +7] and [+16..].
// ---------------------------------------------------------------------------
template <int NT>
__global__ void gemm_bf16_kernel(const bf16_t* __restrict__ X,
                                 const bf16_t* __restrict__ Wt,
                                 float* __restrict__ Y,
                                 int rows, int Kp, int Np,
                                 int colBase, int chunks) {
    const int gw   = (blockIdx.x * blockDim.x + threadIdx.x) >> 5;
    const int nW   = (gridDim.x * blockDim.x) >> 5;
    const int lid  = threadIdx.x & 31;
    const int half = lid >> 4;
    const int l16  = lid & 15;
    const int tiles = (rows >> 4) * chunks;
    for (int t = gw; t < tiles; t += nW) {
        const int row0 = (t / chunks) << 4;
        const int col0 = colBase + (t % chunks) * (NT * 16);
        const bf16_t* arl = X + (size_t)(row0 + l16) * Kp + half * 8;
        const bf16_t* br[NT];
        #pragma unroll
        for (int ci = 0; ci < NT; ++ci)
            br[ci] = Wt + (size_t)(col0 + ci * 16 + l16) * Kp + half * 8;
        v8f acc[NT];
        #pragma unroll
        for (int ci = 0; ci < NT; ++ci) acc[ci] = (v8f){0.f,0.f,0.f,0.f,0.f,0.f,0.f,0.f};
        for (int kb = 0; kb < Kp; kb += 32) {
            v8bf alo = *(const v8bf*)(arl + kb);
            v8bf ahi = *(const v8bf*)(arl + kb + 16);
            v16bf a;
            #pragma unroll
            for (int i = 0; i < 8; ++i) { a[i] = alo[i]; a[i + 8] = ahi[i]; }
            #pragma unroll
            for (int ci = 0; ci < NT; ++ci) {
                v8bf blo = *(const v8bf*)(br[ci] + kb);
                v8bf bhi = *(const v8bf*)(br[ci] + kb + 16);
                v16bf bm;
                #pragma unroll
                for (int i = 0; i < 8; ++i) { bm[i] = blo[i]; bm[i + 8] = bhi[i]; }
                acc[ci] = __builtin_amdgcn_wmma_f32_16x16x32_bf16(
                    false, a, false, bm, (short)0, acc[ci], false, false);
            }
        }
        float* ypb = Y + (size_t)(row0 + half * 8) * Np + col0 + l16;
        #pragma unroll
        for (int ci = 0; ci < NT; ++ci) {
            float* yp = ypb + ci * 16;
            #pragma unroll
            for (int r = 0; r < 8; ++r) yp[(size_t)r * Np] = acc[ci][r];
        }
    }
}

// ---------------------------------------------------------------------------
// Per-channel sum / sumsq over all rows (batch-norm statistics).
// ---------------------------------------------------------------------------
__global__ void colstats_kernel(const float* __restrict__ Y, int rows, int C,
                                float* __restrict__ sums) {
    __shared__ float ssum[1024];
    __shared__ float ssq[1024];
    for (int c = threadIdx.x; c < C; c += blockDim.x) { ssum[c] = 0.f; ssq[c] = 0.f; }
    __syncthreads();
    int rpb = (rows + gridDim.x - 1) / gridDim.x;
    size_t r0 = (size_t)blockIdx.x * rpb;
    size_t r1 = r0 + rpb; if (r1 > (size_t)rows) r1 = rows;
    for (size_t i = r0 * C + threadIdx.x; i < r1 * C; i += blockDim.x) {
        float v = Y[i];
        int c = (int)(i % C);
        atomicAdd(&ssum[c], v);
        atomicAdd(&ssq[c], v * v);
    }
    __syncthreads();
    for (int c = threadIdx.x; c < C; c += blockDim.x) {
        atomicAdd(&sums[c], ssum[c]);
        atomicAdd(&sums[1024 + c], ssq[c]);
    }
}

__global__ void bnfinalize_kernel(const float* __restrict__ sums,
                                  const float* __restrict__ g,
                                  const float* __restrict__ b,
                                  int C, float count, float* __restrict__ ss) {
    int c = threadIdx.x;
    if (c < C) {
        float mean = sums[c] / count;
        float var  = sums[1024 + c] / count - mean * mean;
        float sc   = g[c] * rsqrtf(var + BN_EPS);
        ss[c]        = sc;
        ss[1024 + c] = b[c] - mean * sc;
    }
}

// BN affine + LeakyReLU + re-quantize to bf16 (K zero-padded for next GEMM).
__global__ void bnconv_kernel(const float* __restrict__ Y, int rows, int C, int Kp,
                              const float* __restrict__ ss, bf16_t* __restrict__ XB) {
    size_t n = (size_t)rows * Kp;
    for (size_t idx = blockIdx.x * (size_t)blockDim.x + threadIdx.x; idx < n;
         idx += (size_t)gridDim.x * blockDim.x) {
        int c = (int)(idx % Kp);
        size_t r = idx / Kp;
        float v = 0.f;
        if (c < C) v = lrelu(Y[r * C + c] * ss[c] + ss[1024 + c]);
        XB[idx] = f2bf(v);
    }
}

// BN affine + LeakyReLU + max over the `ns` samples of each group, written
// into the concatenated feature tensor at channel offset coff.
__global__ void maxpool_kernel(const float* __restrict__ Y, int groups, int ns, int C,
                               const float* __restrict__ ss, float* __restrict__ outf,
                               int totalC, int coff) {
    size_t n = (size_t)groups * C;
    for (size_t idx = blockIdx.x * (size_t)blockDim.x + threadIdx.x; idx < n;
         idx += (size_t)gridDim.x * blockDim.x) {
        int c = (int)(idx % C);
        size_t g = idx / C;
        const float* p = Y + (g * (size_t)ns) * C + c;
        float sc = ss[c], sh = ss[1024 + c];
        float m = -3.4e38f;
        for (int j = 0; j < ns; ++j) {
            float t = lrelu(p[(size_t)j * C] * sc + sh);
            m = t > m ? t : m;
        }
        outf[g * totalC + coff + c] = m;
    }
}

// SA3 group-all input: concat(xyz2, feats2) -> bf16 [rows, Kp].
__global__ void sa3build_kernel(const float* __restrict__ nxyz,
                                const float* __restrict__ feats,
                                int rows, int C, int Kp, bf16_t* __restrict__ XB) {
    size_t n = (size_t)rows * Kp;
    for (size_t idx = blockIdx.x * (size_t)blockDim.x + threadIdx.x; idx < n;
         idx += (size_t)gridDim.x * blockDim.x) {
        int c = (int)(idx % Kp);
        size_t r = idx / Kp;
        float v;
        if      (c < 3)     v = nxyz[r * 3 + c];
        else if (c < 3 + C) v = feats[r * C + (c - 3)];
        else                v = 0.f;
        XB[idx] = f2bf(v);
    }
}

// Plain fp32 -> bf16 convert (FC input).
__global__ void conv_kernel(const float* __restrict__ src, int rows, int C, int Kp,
                            bf16_t* __restrict__ XB) {
    size_t n = (size_t)rows * Kp;
    for (size_t idx = blockIdx.x * (size_t)blockDim.x + threadIdx.x; idx < n;
         idx += (size_t)gridDim.x * blockDim.x) {
        int c = (int)(idx % Kp);
        size_t r = idx / Kp;
        float v = (c < C) ? src[r * C + c] : 0.f;
        XB[idx] = f2bf(v);
    }
}

// Final 16x64 * 64x1 + bias.
__global__ void fc_final_kernel(const bf16_t* __restrict__ X,
                                const float* __restrict__ W4,
                                const float* __restrict__ b4,
                                float* __restrict__ out) {
    int i = threadIdx.x;
    if (i < 16) {
        float a = b4[0];
        for (int k = 0; k < 64; ++k) a += bf2f(X[i * 64 + k]) * W4[k];
        out[i] = a;
    }
}

// ---------------------------------------------------------------------------
// Host orchestration.  Input order (setup_inputs insertion order, pytree
// leaves W,g,b per layer):
//   [0] pointcloud (16,8192,3)
//   [1..27]  sa1: 3 scales x 3 layers x (W,g,b)
//   [28..54] sa2: 3 scales x 3 layers x (W,g,b)
//   [55..63] sa3: 3 layers x (W,g,b)
//   [64..74] fc:  W1,g1,b1,W2,g2,b2,W3,g3,b3,W4,b4
// ---------------------------------------------------------------------------
extern "C" void kernel_launch(void* const* d_in, const int* in_sizes, int n_in,
                              void* d_out, int out_size, void* d_ws, size_t ws_size,
                              hipStream_t stream) {
    (void)in_sizes; (void)n_in; (void)out_size; (void)ws_size;
    const float* pc = (const float*)d_in[0];
    auto P = [&](int i) -> const float* { return (const float*)d_in[i]; };

    char* ws = (char*)d_ws;
    size_t off = 0;
    auto alloc = [&](size_t bytes) -> char* {
        char* p = ws + off;
        off += (bytes + 255) & ~(size_t)255;
        return p;
    };
    float*  Yf     = (float*) alloc((size_t)134217728 * 4); // fp32 GEMM out (max 1M x 128)
    bf16_t* XB     = (bf16_t*)alloc((size_t)100663296 * 2); // bf16 activations (max 1M x 96)
    bf16_t* WB     = (bf16_t*)alloc((size_t)1048576 * 2);   // bf16 weights (max 1024x512/512x1024)
    float*  feats1 = (float*) alloc((size_t)16 * 512 * 320 * 4);
    float*  feats2 = (float*) alloc((size_t)16 * 128 * 640 * 4);
    float*  nxyz1  = (float*) alloc((size_t)16 * 512 * 3 * 4);
    float*  nxyz2  = (float*) alloc((size_t)16 * 128 * 3 * 4);
    int*    fidx1  = (int*)   alloc((size_t)16 * 512 * 4);
    int*    fidx2  = (int*)   alloc((size_t)16 * 128 * 4);
    float*  sa3out = (float*) alloc((size_t)16 * 1024 * 4);
    float*  sums   = (float*) alloc(2048 * 4);
    float*  ss     = (float*) alloc(2048 * 4);

    auto pad32 = [](int x) { return (x + 31) & ~31; };
    auto nb_for = [](size_t n) -> int {
        size_t b = (n + 255) / 256;
        if (b < 1) b = 1;
        if (b > 65536) b = 65536;
        return (int)b;
    };
    auto blocks_for_tiles = [](int tiles) -> int {
        int blocks = (tiles + 7) / 8;
        if (blocks > 16384) blocks = 16384;
        if (blocks < 1) blocks = 1;
        return blocks;
    };

    auto run_wconv = [&](const float* Wf, int Cout, int Cin, int Kp) {
        size_t n = (size_t)Cout * Kp;
        wconv_kernel<<<nb_for(n), 256, 0, stream>>>(Wf, Cout, Cin, Kp, WB);
    };
    auto run_gemm = [&](int rows, int Kp, int Np) {
        int colT = Np / 16;          // even (Np is a multiple of 32)
        int c4   = colT / 4;         // 64-wide chunks
        int rem  = colT - c4 * 4;    // 0 or 2
        if (c4 > 0) {
            int tiles = (rows / 16) * c4;
            gemm_bf16_kernel<4><<<blocks_for_tiles(tiles), 256, 0, stream>>>(
                XB, WB, Yf, rows, Kp, Np, 0, c4);
        }
        if (rem == 2) {
            int tiles = rows / 16;
            gemm_bf16_kernel<2><<<blocks_for_tiles(tiles), 256, 0, stream>>>(
                XB, WB, Yf, rows, Kp, Np, c4 * 64, 1);
        }
    };
    auto run_stats = [&](int rows, int C, const float* g, const float* b) {
        hipMemsetAsync(sums, 0, 2048 * sizeof(float), stream);
        int nb = (rows + 255) / 256;
        if (nb > 1024) nb = 1024;
        if (nb < 1) nb = 1;
        colstats_kernel<<<nb, 256, 0, stream>>>(Yf, rows, C, sums);
        bnfinalize_kernel<<<1, 1024, 0, stream>>>(sums, g, b, C, (float)rows, ss);
    };
    auto run_layer = [&](int rows, int Cin, int Kp, int Cout, int widx) {
        run_wconv(P(widx), Cout, Cin, Kp);
        run_gemm(rows, Kp, Cout);
        run_stats(rows, Cout, P(widx + 1), P(widx + 2));
    };
    auto run_bnconv = [&](int rows, int C, int KpN) {
        size_t n = (size_t)rows * KpN;
        bnconv_kernel<<<nb_for(n), 256, 0, stream>>>(Yf, rows, C, KpN, ss, XB);
    };
    auto run_mlp = [&](int rows, const int* d, int widx) {
        run_layer(rows, d[0], pad32(d[0]), d[1], widx);
        run_bnconv(rows, d[1], pad32(d[1]));
        run_layer(rows, d[1], pad32(d[1]), d[2], widx + 3);
        run_bnconv(rows, d[2], pad32(d[2]));
        run_layer(rows, d[2], pad32(d[2]), d[3], widx + 6);
    };
    auto run_maxpool = [&](int groups, int nsmp, int C, float* outf, int totalC, int coff) {
        size_t n = (size_t)groups * C;
        maxpool_kernel<<<nb_for(n), 256, 0, stream>>>(Yf, groups, nsmp, C, ss, outf, totalC, coff);
    };

    // ---------------- SA1 ----------------
    fps_kernel<<<16, 256, 8192 * sizeof(float), stream>>>(pc, 8192, 512, fidx1, nxyz1);
    {
        const int   ns1[3]   = {16, 32, 128};
        const float r1[3]    = {0.1f, 0.2f, 0.4f};
        const int   d1[3][4] = {{6, 32, 32, 64}, {6, 64, 64, 128}, {6, 64, 96, 128}};
        const int   co1[3]   = {0, 64, 192};
        for (int i = 0; i < 3; ++i) {
            group_kernel<<<(16 * 512 + 7) / 8, 256, 0, stream>>>(
                pc, pc, 3, nxyz1, 8192, 512, 16 * 512, ns1[i], r1[i] * r1[i], 32, XB);
            int rows = 16 * 512 * ns1[i];
            run_mlp(rows, d1[i], 1 + i * 9);
            run_maxpool(16 * 512, ns1[i], d1[i][3], feats1, 320, co1[i]);
        }
    }
    // ---------------- SA2 ----------------
    fps_kernel<<<16, 256, 512 * sizeof(float), stream>>>(nxyz1, 512, 128, fidx2, nxyz2);
    {
        const int   ns2[3]    = {32, 64, 128};
        const float r2a[3]    = {0.2f, 0.4f, 0.8f};
        const int   d2c[3][4] = {{323, 64, 64, 128}, {323, 128, 128, 256}, {323, 128, 128, 256}};
        const int   co2[3]    = {0, 128, 384};
        for (int i = 0; i < 3; ++i) {
            group_kernel<<<(16 * 128 + 7) / 8, 256, 0, stream>>>(
                nxyz1, feats1, 320, nxyz2, 512, 128, 16 * 128, ns2[i], r2a[i] * r2a[i], 352, XB);
            int rows = 16 * 128 * ns2[i];
            run_mlp(rows, d2c[i], 28 + i * 9);
            run_maxpool(16 * 128, ns2[i], d2c[i][3], feats2, 640, co2[i]);
        }
    }
    // ---------------- SA3 (group-all) ----------------
    {
        size_t n = (size_t)2048 * 672;
        sa3build_kernel<<<nb_for(n), 256, 0, stream>>>(nxyz2, feats2, 2048, 640, 672, XB);
        const int d3[4] = {643, 256, 512, 1024};
        run_mlp(2048, d3, 55);
        run_maxpool(16, 128, 1024, sa3out, 1024, 0);
    }
    // ---------------- FC head ----------------
    {
        size_t n = (size_t)16 * 1024;
        conv_kernel<<<nb_for(n), 256, 0, stream>>>(sa3out, 16, 1024, 1024, XB);
        run_layer(16, 1024, 1024, 512, 64); run_bnconv(16, 512, 512);
        run_layer(16, 512, 512, 256, 67);   run_bnconv(16, 256, 256);
        run_layer(16, 256, 256, 64, 70);    run_bnconv(16, 64, 64);
        fc_final_kernel<<<1, 16, 0, stream>>>(XB, P(73), P(74), (float*)d_out);
    }
}
